// TemporalAttentionLayer_65292092833944
// MI455X (gfx1250) — matrix operand order repaired and checked by reading, not compiled, not run on hardware
//
#include <hip/hip_runtime.h>
#include <cstdint>

#define LRELU_ALPHA 0.2f
constexpr int Bn = 64;
constexpr int Kn = 1024;
constexpr int Nn = 128;
constexpr int TM = 128;     // output rows per block (8 waves x 16-row strips)
constexpr int JSTRIDE = 40; // LDS row stride in halves (multiple of 8 -> b128-aligned)

typedef __attribute__((ext_vector_type(16))) _Float16 v16h;
typedef __attribute__((ext_vector_type(8)))  _Float16 v8h;
typedef __attribute__((ext_vector_type(8)))  float    v8f;

// ---------------------------------------------------------------------------
// Kernel 1: s1[b,k] = x[b,k,:] . w[0:128],  s2[b,k] = x[b,k,:] . w[128:256]
// One wave per row; float4 loads, wave32 shuffle reduction.
// ---------------------------------------------------------------------------
__global__ __launch_bounds__(256) void ta_scores(const float* __restrict__ x,
                                                 const float* __restrict__ w,
                                                 float* __restrict__ s1,
                                                 float* __restrict__ s2) {
  const int lane = threadIdx.x & 31;
  const int wv   = threadIdx.x >> 5;
  const long row = (long)blockIdx.x * 8 + wv;            // 0 .. B*K-1
  const float4 xv = ((const float4*)(x + row * Nn))[lane];
  const float4 w1 = ((const float4*)w)[lane];
  const float4 w2 = ((const float4*)(w + Nn))[lane];
  float d1 = xv.x * w1.x + xv.y * w1.y + xv.z * w1.z + xv.w * w1.w;
  float d2 = xv.x * w2.x + xv.y * w2.y + xv.z * w2.z + xv.w * w2.w;
#pragma unroll
  for (int off = 16; off >= 1; off >>= 1) {
    d1 += __shfl_xor(d1, off, 32);
    d2 += __shfl_xor(d2, off, 32);
  }
  if (lane == 0) {
    s1[row] = d1;
    s2[row] = d2;
  }
}

// ---------------------------------------------------------------------------
// Kernel 2: s2max[b] = max_k s2[b,k]
// ---------------------------------------------------------------------------
__global__ __launch_bounds__(256) void ta_s2max(const float* __restrict__ s2,
                                                float* __restrict__ s2max) {
  __shared__ float red[256];
  const int b = blockIdx.x;
  float m = -3.402823466e38f;
  for (int k = threadIdx.x; k < Kn; k += 256) m = fmaxf(m, s2[b * Kn + k]);
  red[threadIdx.x] = m;
  __syncthreads();
  for (int s = 128; s > 0; s >>= 1) {
    if (threadIdx.x < s) red[threadIdx.x] = fmaxf(red[threadIdx.x], red[threadIdx.x + s]);
    __syncthreads();
  }
  if (threadIdx.x == 0) s2max[b] = red[0];
}

// ---------------------------------------------------------------------------
// Kernel 3: fused  h[b,i,:] = softmax_j(lrelu(s1_i+s2_j)) @ x[b,:,:]
// 256 threads (8 waves). Block tile 128 x 128: each wave owns one 16-row
// strip (A fragment / exp computed exactly once) and all 8 n-tiles
// (8 WMMAs per wave per K-step of 32). Double-buffered transposed f16 LDS.
// ---------------------------------------------------------------------------
__global__ __launch_bounds__(256) void ta_attn(const float* __restrict__ x,
                                               const float* __restrict__ s1,
                                               const float* __restrict__ s2,
                                               const float* __restrict__ s2max,
                                               float* __restrict__ out) {
  const int b    = blockIdx.y;
  const int i0   = blockIdx.x * TM;
  const int tid  = threadIdx.x;
  const int wave = tid >> 5;             // 0..7 : this wave's 16-row strip
  const int lane = tid & 31;
  const int hi   = (lane >= 16) ? 1 : 0;

  __shared__ alignas(16) _Float16 xs[2][Nn][JSTRIDE]; // transposed tile, 2x10KB
  __shared__ float s2s[2][32];
  __shared__ float s1s[TM];

  if (tid < TM) s1s[tid] = s1[b * Kn + i0 + tid];
  __syncthreads();

  const float s1v = s1s[wave * 16 + (lane & 15)];
  float rowmax;
  {
    float t = s1v + s2max[b];
    rowmax = (t > 0.f) ? t : LRELU_ALPHA * t;
  }

  v8f acc[8] = {v8f{}, v8f{}, v8f{}, v8f{}, v8f{}, v8f{}, v8f{}, v8f{}};
  float dsum = 0.f;

  const float* __restrict__ xb  = x + (size_t)b * Kn * Nn;
  const float* __restrict__ s2b = s2 + (size_t)b * Kn;

  const int nc = tid & 127;        // staging: column owned by this thread
  const int jb = (tid >> 7) * 8;   // staging: first j-row of each 8-run

  // stage tile j0 into buffer `buf` (global f32 -> f16, transposed, b128 stores)
  auto stage = [&](int j0, int buf) {
    if (tid < 32) s2s[buf][tid] = s2b[j0 + tid];
#pragma unroll
    for (int g = 0; g < 2; ++g) {
      const int js = jb + g * 16;          // 0/8 or 16/24
      float f[8];
#pragma unroll
      for (int u = 0; u < 8; ++u)
        f[u] = xb[(size_t)(j0 + js + u) * Nn + nc];
      v8h h8;
#pragma unroll
      for (int u = 0; u < 8; ++u) h8[u] = (_Float16)f[u];
      *(v8h*)&xs[buf][nc][js] = h8;
    }
  };

  stage(0, 0);

  for (int j0 = 0; j0 < Kn; j0 += 32) {
    const int buf = (j0 >> 5) & 1;
    __syncthreads();                       // buf ready; buf^1 free for refill

    if (j0 + 32 < Kn) stage(j0 + 32, buf ^ 1);
    if (j0 + 64 < Kn)
      __builtin_prefetch(&xb[(size_t)(j0 + 64) * Nn + (tid << 4)], 0, 0);

    // ---- A fragment: p_ij = exp(lrelu(s1_i + s2_j) - rowmax_i), f16 ----
    // A k order used here (contiguous): lane<16: K in {0..15}; lane>=16: {16..31}
    v16h afrag;
#pragma unroll
    for (int h = 0; h < 16; ++h) {
      const int v = h >> 1;
      const int k = (hi ? 16 : 0) + h + 0 * v;
      float e = s1v + s2s[buf][k];
      e = (e > 0.f) ? e : LRELU_ALPHA * e;
      float p = __expf(e - rowmax);
      dsum += p;
      afrag[h] = (_Float16)p;
    }

    // ---- B fragments: 2x ds_load_b128 each; 8 WMMAs (all n-tiles) ----
    // B k order used here, same as A: lanes 0-15 hold K=0..15, lanes 16-31 hold K=16..31
    const int kb0 = hi ? 16 : 0;
#pragma unroll
    for (int t = 0; t < 8; ++t) {
      const int n = t * 16 + (lane & 15);
      const v8h* bp = (const v8h*)&xs[buf][n][kb0];
      const v8h blo = bp[0];
      const v8h bhi = bp[1];
      const v16h bfrag = __builtin_shufflevector(
          blo, bhi, 0, 1, 2, 3, 4, 5, 6, 7, 8, 9, 10, 11, 12, 13, 14, 15);
      acc[t] = __builtin_amdgcn_wmma_f32_16x16x32_f16(
          false, afrag, false, bfrag, (short)0, acc[t], false, false);
    }
  }

  // ---- denominator: lanes l and l^16 jointly cover all 32 K per row ----
  dsum += __shfl_xor(dsum, 16, 32);       // lane m (and m+16) hold denom[m]
  float inv[8];
#pragma unroll
  for (int r = 0; r < 8; ++r) {
    const int m = r + (hi ? 8 : 0);       // C/D layout: VGPR r -> row r (+8 hi)
    inv[r] = 1.0f / __shfl(dsum, m, 32);
  }

  // ---- normalize + store (each store: 2 x 64B contiguous segments) ----
  float* __restrict__ ob =
      out + (size_t)b * Kn * Nn + (size_t)(i0 + wave * 16) * Nn;
#pragma unroll
  for (int t = 0; t < 8; ++t) {
    const int n = t * 16 + (lane & 15);
#pragma unroll
    for (int r = 0; r < 8; ++r) {
      const int m = r + (hi ? 8 : 0);
      ob[(size_t)m * Nn + n] = acc[t][r] * inv[r];
    }
  }
}

// ---------------------------------------------------------------------------
extern "C" void kernel_launch(void* const* d_in, const int* in_sizes, int n_in,
                              void* d_out, int out_size, void* d_ws, size_t ws_size,
                              hipStream_t stream) {
  const float* x = (const float*)d_in[0];   // (64, 1024, 128) f32
  const float* w = (const float*)d_in[1];   // (256, 1) f32
  float* out = (float*)d_out;               // (64, 1024, 128) f32

  float* s1  = (float*)d_ws;                // B*K
  float* s2  = s1 + Bn * Kn;                // B*K
  float* s2m = s2 + Bn * Kn;                // B

  ta_scores<<<Bn * Kn / 8, 256, 0, stream>>>(x, w, s1, s2);
  ta_s2max<<<Bn, 256, 0, stream>>>(s2, s2m);
  dim3 grid(Kn / TM, Bn);
  ta_attn<<<grid, 256, 0, stream>>>(x, s1, s2, s2m, out);
}
